// Attention1_21397527069263
// MI455X (gfx1250) — matrix operand order repaired and checked by reading, not compiled, not run on hardware
//
#include <hip/hip_runtime.h>
#include <hip/hip_bf16.h>

// Problem constants (from reference): B=8, S=4096, D=256
#define BB 8
#define SS 4096
#define DD 256

typedef __attribute__((ext_vector_type(16))) __bf16 v16bf;
typedef __attribute__((ext_vector_type(8)))  float  v8f;

__device__ __forceinline__ unsigned short f2bfu(float f) {
    return __builtin_bit_cast(unsigned short, (__bf16)f);   // v_cvt_bf16 (RNE)
}
__device__ __forceinline__ float bf2f(unsigned short s) {
    return __builtin_bit_cast(float, ((unsigned)s) << 16);
}

// Prep: x (fp32) -> xbf (bf16), row-major [B,S,D]. One pass, bandwidth-trivial.
__global__ __launch_bounds__(256)
void cvt_bf16_kernel(const float* __restrict__ x, __bf16* __restrict__ xbf) {
    size_t i = ((size_t)blockIdx.x * 256 + threadIdx.x) * 8;
    #pragma unroll
    for (int e = 0; e < 8; ++e) xbf[i + e] = (__bf16)x[i + e];
}

// ---- helpers: 8-fragment B load, 8-deep WMMA chain, score-tile store ----
__device__ __forceinline__ void loadB(v16bf dst[8], const __bf16* __restrict__ xkrow, int h) {
    #pragma unroll
    for (int c = 0; c < 8; ++c)
        dst[c] = *(const v16bf*)(xkrow + 32 * c + 16 * h);   // 32B-aligned vector load
}
__device__ __forceinline__ v8f wmma8(const v16bf a[8], const v16bf bfr[8]) {
    v8f acc = {};
    #pragma unroll
    for (int c = 0; c < 8; ++c)
        acc = __builtin_amdgcn_wmma_f32_16x16x32_bf16(
            false, a[c], false, bfr[c], (short)0, acc, false, false);
    return acc;
}
__device__ __forceinline__ void storeScores(unsigned short* __restrict__ sc,
                                            v8f acc, int row0, int col) {
    #pragma unroll
    for (int r = 0; r < 8; ++r)
        sc[(row0 + r) * SS + col] = f2bfu(acc[r]);
}

// Kernel 1: per (b, 32-row q-block): scores via WMMA (register double-buffered B),
// single-pass softmax over a 256KB LDS score block, one atomic per (WG, column)
// into w[b,k].  PRECONV=true: B fragments come from pre-converted bf16.
template <bool PRECONV>
__global__ __launch_bounds__(256)
void attn_scores_softmax_kernel(const float* __restrict__ x,
                                const float* __restrict__ mask,
                                const __bf16* __restrict__ xbf,
                                float* __restrict__ w) {
    __shared__ unsigned short sc[32 * SS];   // 256 KB bf16 score block
    __shared__ float sm_m[32];
    __shared__ float sm_iz[32];

    const int b     = blockIdx.x >> 7;          // 128 q-blocks per batch
    const int qBase = (blockIdx.x & 127) * 32;

    const int tid  = threadIdx.x;
    const int wave = tid >> 5;                  // 0..7
    const int lane = tid & 31;
    const int i    = wave >> 2;                 // row tile 0..1 (16 rows each)
    const int j    = wave & 3;                  // col tile 0..3 (16 cols each)
    const int h    = lane >> 4;                 // half-wave 0/1
    const int m    = lane & 15;

    const float*  xb  = x    + (size_t)b * SS * DD;
    const float*  mb  = mask + (size_t)b * SS * DD;
    const __bf16* xbb = PRECONV ? (xbf + (size_t)b * SS * DD) : nullptr;

    // ---- Load A fragments once: query rows (x*mask*1/sqrt(D)) -> bf16 ----
    // 16-bit A 16x32, k order used here (B matches): lanes 0-15 hold K=0..15; lanes 16-31 hold K=16..31
    v16bf afrag[8];
    {
        const int qr = qBase + 16 * i + m;
        const float* xr = xb + (size_t)qr * DD;
        const float* mr = mb + (size_t)qr * DD;
        #pragma unroll
        for (int c = 0; c < 8; ++c) {
            const int d0 = 32 * c + 16 * h;
            #pragma unroll
            for (int e = 0; e < 8; ++e) {
                afrag[c][e]     = (__bf16)(xr[d0 + e]      * mr[d0 + e]      * 0.0625f);
                afrag[c][e + 8] = (__bf16)(xr[d0 + 8 + e] * mr[d0 + 8 + e] * 0.0625f);
            }
        }
    }

    const int row0   = 16 * i + 8 * h;          // f32 C/D tile: VGPR r -> row row0+r, col m
    const int colOff = 16 * j + m;              // this lane's B column / score column

    if (PRECONV) {
        // ---- k loop, unrolled x2 with register double-buffered B fragments ----
        v16bf b0[8], b1[8];
        loadB(b0, xbb + (size_t)colOff * DD, h);
        for (int kt = 0; kt < SS; kt += 128) {
            const int k1 = kt + 64;
            loadB(b1, xbb + (size_t)(k1 + colOff) * DD, h);          // prefetch tile n+1
            storeScores(sc, wmma8(afrag, b0), row0, kt + colOff);    // compute tile n
            const int k2 = (kt + 128) & (SS - 1);                    // wraps on last iter
            loadB(b0, xbb + (size_t)(k2 + colOff) * DD, h);          // prefetch tile n+2
            storeScores(sc, wmma8(afrag, b1), row0, k1 + colOff);    // compute tile n+1
        }
    } else {
        for (int kt = 0; kt < SS; kt += 64) {
            const int kcol = kt + colOff;
            const float* xk = xb + (size_t)kcol * DD;
            v8f acc = {};
            #pragma unroll
            for (int c = 0; c < 8; ++c) {
                const int d0 = 32 * c + 16 * h;
                v16bf bfrag;
                #pragma unroll
                for (int e = 0; e < 16; ++e) bfrag[e] = (__bf16)xk[d0 + e];
                acc = __builtin_amdgcn_wmma_f32_16x16x32_bf16(
                    false, afrag[c], false, bfrag, (short)0, acc, false, false);
            }
            storeScores(sc, acc, row0, kcol);
        }
    }
    __syncthreads();

    // ---- Per-row softmax stats: each wave handles 4 rows ----
    for (int rr = 0; rr < 4; ++rr) {
        const int row = 4 * wave + rr;
        const unsigned short* srow = sc + row * SS;
        float mx = -3.0e38f;
        for (int k = lane; k < SS; k += 32) mx = fmaxf(mx, bf2f(srow[k]));
        #pragma unroll
        for (int off = 16; off; off >>= 1) mx = fmaxf(mx, __shfl_xor(mx, off, 32));
        float sum = 0.f;
        for (int k = lane; k < SS; k += 32) sum += __expf(bf2f(srow[k]) - mx);
        #pragma unroll
        for (int off = 16; off; off >>= 1) sum += __shfl_xor(sum, off, 32);
        if (lane == 0) { sm_m[row] = mx; sm_iz[row] = 1.0f / sum; }
    }
    __syncthreads();

    // ---- Column sweep: w[b,k] += sum_r exp(s[r,k]-m_r) / Z_r ----
    for (int k = tid; k < SS; k += 256) {
        float acc = 0.f;
        #pragma unroll 8
        for (int r = 0; r < 32; ++r)
            acc += __expf(bf2f(sc[r * SS + k]) - sm_m[r]) * sm_iz[r];
        atomicAdd(&w[(size_t)b * SS + k], acc);
    }
}

// Kernel 2: out[b,:] += w[b,kc]^T @ x[b,kc,:], k split 16-ways for parallelism.
__global__ __launch_bounds__(256)
void attn_out_kernel(const float* __restrict__ x,
                     const float* __restrict__ w,
                     float* __restrict__ out) {
    const int b  = blockIdx.x >> 4;             // 16 k-chunks per batch
    const int kc = (blockIdx.x & 15) * (SS / 16);
    const int d  = threadIdx.x;
    const float* xb = x + (size_t)b * SS * DD;
    const float* wb = w + (size_t)b * SS;
    float acc = 0.f;
    #pragma unroll 4
    for (int k = kc; k < kc + SS / 16; ++k)
        acc = fmaf(wb[k], xb[(size_t)k * DD + d], acc);
    atomicAdd(&out[b * DD + d], acc);
}

extern "C" void kernel_launch(void* const* d_in, const int* in_sizes, int n_in,
                              void* d_out, int out_size, void* d_ws, size_t ws_size,
                              hipStream_t stream) {
    const float* x    = (const float*)d_in[0];
    const float* mask = (const float*)d_in[1];
    float* out = (float*)d_out;

    const size_t wBytes   = (size_t)BB * SS * sizeof(float);        // 128 KB
    const size_t xbfBytes = (size_t)BB * SS * DD * sizeof(__bf16);  // 16 MB

    float*  w   = (float*)d_ws;
    __bf16* xbf = (__bf16*)((char*)d_ws + wBytes);                  // 128KB-aligned

    hipMemsetAsync(w, 0, wBytes, stream);
    hipMemsetAsync(out, 0, (size_t)BB * DD * sizeof(float), stream);

    if (ws_size >= wBytes + xbfBytes) {
        const size_t nElem = (size_t)BB * SS * DD;                  // 8.4M
        cvt_bf16_kernel<<<dim3((unsigned)(nElem / (256 * 8))), dim3(256), 0, stream>>>(x, xbf);
        attn_scores_softmax_kernel<true>
            <<<dim3(BB * (SS / 32)), dim3(256), 0, stream>>>(x, mask, xbf, w);
    } else {
        attn_scores_softmax_kernel<false>
            <<<dim3(BB * (SS / 32)), dim3(256), 0, stream>>>(x, mask, nullptr, w);
    }
    attn_out_kernel<<<dim3(BB * 16), dim3(256), 0, stream>>>(x, w, out);
}
